// SpconvMiddleExtractor_29420525977722
// MI455X (gfx1250) — matrix operand order repaired and checked by reading, not compiled, not run on hardware
//
#include <hip/hip_runtime.h>
#include <hip/hip_bf16.h>

typedef __bf16 bf16_t;
typedef __attribute__((ext_vector_type(16))) __bf16 v16bf;
typedef __attribute__((ext_vector_type(8)))  __bf16 v8bf;
typedef __attribute__((ext_vector_type(8)))  float  v8f;

#define BN_EPS 1e-3f

static __device__ __forceinline__ v16bf cat16(v8bf lo, v8bf hi) {
  return __builtin_shufflevector(lo, hi, 0, 1, 2, 3, 4, 5, 6, 7,
                                 8, 9, 10, 11, 12, 13, 14, 15);
}

// ---------------- prep kernels ----------------

__global__ void k_mask_thresh(const float* __restrict__ u, float* __restrict__ m, int n) {
  int i = blockIdx.x * 256 + threadIdx.x;
  if (i < n) m[i] = (u[i] < 0.05f) ? 1.0f : 0.0f;
}

// mo[od][h][w] = any(mi[2*od + kd][h][w]) for kd in 0..2   (VALID, stride 2)
__global__ void k_mask_down(const float* __restrict__ mi, float* __restrict__ mo,
                            int Dout, int HW) {
  int i = blockIdx.x * 256 + threadIdx.x;
  int n = Dout * HW;
  if (i >= n) return;
  int od = i / HW, hw = i - od * HW;
  float s = mi[(2 * od) * HW + hw] + mi[(2 * od + 1) * HW + hw] + mi[(2 * od + 2) * HW + hw];
  mo[i] = (s > 0.0f) ? 1.0f : 0.0f;
}

// fold BN: y = x*s + t ; s = g*rsqrt(v+eps), t = b - mu*s   (5 layers x 64 ch)
__global__ void k_bnfold(const float* __restrict__ g, const float* __restrict__ b,
                         const float* __restrict__ mu, const float* __restrict__ v,
                         float* __restrict__ s, float* __restrict__ t) {
  int i = blockIdx.x * 64 + threadIdx.x;
  if (i < 320) {
    float sc = g[i] * rsqrtf(v[i] + BN_EPS);
    s[i] = sc;
    t[i] = b[i] - mu[i] * sc;
  }
}

// w[co][ci][taps...] fp32  ->  wt[tap][co][ci] bf16
__global__ void k_wrepack(const float* __restrict__ w, bf16_t* __restrict__ wt,
                          int CO, int CIN, int TAPS) {
  int i = blockIdx.x * 256 + threadIdx.x;
  int n = TAPS * CO * CIN;
  if (i >= n) return;
  int tap = i / (CO * CIN);
  int rem = i - tap * CO * CIN;
  int co = rem / CIN, ci = rem - co * CIN;
  wt[i] = (bf16_t)w[(co * CIN + ci) * TAPS + tap];
}

// x [C][D][H][W] fp32 -> xb [D][H][W][C] bf16, with mask applied (LDS transpose)
__global__ void k_x_to_dhwc(const float* __restrict__ x, const float* __restrict__ mask,
                            bf16_t* __restrict__ xb, int CIN, int W) {
  __shared__ float tile[16][17];
  int w0 = blockIdx.x * 16;
  int dh = blockIdx.y;            // fused d*H + h, gridDim.y == D*H
  int c0 = blockIdx.z * 16;
  int tx = threadIdx.x, ty = threadIdx.y;
  size_t DHW = (size_t)gridDim.y * W;
  tile[ty][tx] = x[(size_t)(c0 + ty) * DHW + (size_t)dh * W + w0 + tx];
  __syncthreads();
  int w = w0 + ty;
  float mk = mask[(size_t)dh * W + w];
  xb[((size_t)dh * W + w) * CIN + c0 + tx] = (bf16_t)(tile[tx][ty] * mk);
}

// ---------------- implicit-GEMM conv + BN + ReLU + mask (WMMA bf16) ----------------
// One wave (32 threads) computes a 64(M=Cout) x 16(N=w-tile) x 2(h-rows) output block.
// in : [Din][H][W][CIN] bf16 (channels-last), wt : [tap][64][CIN] bf16.
// SUBM: 3x3x3 SAME, Dout==Din.  !SUBM: (3,1,1) stride (2,1,1) VALID (no bounds checks).

template <int CIN, bool SUBM>
__global__ void __launch_bounds__(32)
k_conv_wmma(const bf16_t* __restrict__ in, const bf16_t* __restrict__ wt,
            const float* __restrict__ scale, const float* __restrict__ shift,
            const float* __restrict__ maskO,
            bf16_t* __restrict__ outb, float* __restrict__ outf,
            int Din, int Dout, int H, int W) {
  const int lane  = threadIdx.x;
  const int col   = lane & 15;     // N column within tile
  const int hi    = lane >> 4;     // lane-half select
  const int wbase = blockIdx.x * 16;
  const int oh0   = blockIdx.y * 2;
  const int od    = blockIdx.z;

  v8f acc[2][4];
#pragma unroll
  for (int r = 0; r < 2; ++r)
#pragma unroll
    for (int m = 0; m < 4; ++m)
#pragma unroll
      for (int j = 0; j < 8; ++j) acc[r][m][j] = 0.0f;

  const int KH  = SUBM ? 3 : 1;
  const int KW  = SUBM ? 3 : 1;
  const int aks = hi ? 8 : 0;      // A: lanes 0-15 -> K {0..7,16..23}; 16-31 -> {8..15,24..31}
  const int bks = hi ? 8 : 0;      // B: lanes 0-15 -> K {0..7,16..23}; 16-31 -> {8..15,24..31}

  v16bf bzero;
#pragma unroll
  for (int j = 0; j < 16; ++j) bzero[j] = (bf16_t)0.0f;

  for (int kd = 0; kd < 3; ++kd) {
    const int dd = SUBM ? (od + kd - 1) : (2 * od + kd);
    if (SUBM && (dd < 0 || dd >= Din)) continue;        // wave-uniform
#pragma unroll
    for (int kh = 0; kh < KH; ++kh)
#pragma unroll
      for (int kw = 0; kw < KW; ++kw) {
        const int tap = (kd * KH + kh) * KW + kw;
        const bf16_t* wt_tap = wt + (size_t)tap * 64 * CIN;
        // keep the (L2-resident) weight slab for the next tap warm in WGP$
        __builtin_prefetch(wt_tap + (size_t)64 * CIN + (size_t)col * CIN, 0, 3);
#pragma unroll
        for (int c0 = 0; c0 < CIN; c0 += 32) {
          // A fragments: 4 M-tiles of 16 output channels, 2 x b128 each
          v16bf afr[4];
#pragma unroll
          for (int m = 0; m < 4; ++m) {
            const bf16_t* ap = wt_tap + (size_t)(m * 16 + col) * CIN + c0 + aks;
            afr[m] = cat16(*(const v8bf*)ap, *(const v8bf*)(ap + 16));
          }
#pragma unroll
          for (int r = 0; r < 2; ++r) {
            const int hh = oh0 + r + (SUBM ? kh - 1 : 0);
            if (SUBM && ((unsigned)hh >= (unsigned)H)) continue;   // wave-uniform
            const int ww = wbase + col + (SUBM ? kw - 1 : 0);
            const bool valid = !SUBM || ((unsigned)ww < (unsigned)W);
            v16bf bfr;
            if (valid) {
              const bf16_t* bp = in + (((size_t)dd * H + hh) * W + ww) * CIN + c0 + bks;
              bfr = cat16(*(const v8bf*)bp, *(const v8bf*)(bp + 16));
            } else {
              bfr = bzero;
            }
#pragma unroll
            for (int m = 0; m < 4; ++m)
              acc[r][m] = __builtin_amdgcn_wmma_f32_16x16x32_bf16(
                  false, afr[m], false, bfr, (short)0, acc[r][m], false, false);
          }
        }
      }
  }

  // epilogue: BN (folded) + ReLU + mask; bf16 channels-last and/or fp32 NCDHW
  const int HW = H * W;
#pragma unroll
  for (int r = 0; r < 2; ++r) {
    const int oh = oh0 + r;
    const float mk = maskO[(size_t)od * HW + (size_t)oh * W + wbase + col];
#pragma unroll
    for (int m = 0; m < 4; ++m) {
      const int cbase = m * 16 + hi * 8;   // D layout: vgpr j -> M = j + 8*hi
      float o[8];
#pragma unroll
      for (int j = 0; j < 8; ++j) {
        float y = acc[r][m][j] * scale[cbase + j] + shift[cbase + j];
        y = y > 0.0f ? y : 0.0f;
        o[j] = y * mk;
      }
      if (outb) {
        v8bf ob;
#pragma unroll
        for (int j = 0; j < 8; ++j) ob[j] = (bf16_t)o[j];
        *(v8bf*)(outb + ((size_t)od * HW + (size_t)oh * W + wbase + col) * 64 + cbase) = ob;
      }
      if (outf) {
#pragma unroll
        for (int j = 0; j < 8; ++j)
          outf[(size_t)(cbase + j) * Dout * HW + (size_t)od * HW + (size_t)oh * W + wbase + col] = o[j];
      }
    }
  }
}

// ---------------- host launcher ----------------

extern "C" void kernel_launch(void* const* d_in, const int* in_sizes, int n_in,
                              void* d_out, int out_size, void* d_ws, size_t ws_size,
                              hipStream_t stream) {
  const float* x      = (const float*)d_in[0];
  const float* mask_u = (const float*)d_in[1];
  const float* w1     = (const float*)d_in[2];   // [64][128][3][3][3]
  const float* w2     = (const float*)d_in[3];   // [64][64][3][1][1]
  const float* w3     = (const float*)d_in[4];   // [64][64][3][3][3]
  const float* w4     = (const float*)d_in[5];
  const float* w5     = (const float*)d_in[6];
  const float* bng    = (const float*)d_in[7];
  const float* bnb    = (const float*)d_in[8];
  const float* bnm    = (const float*)d_in[9];
  const float* bnv    = (const float*)d_in[10];
  float* out = (float*)d_out;

  const int D = 11, H = 200, W = 176, C = 128;
  const int HW = H * W;
  const int DHW = D * HW;
  const int D2 = 5, D3 = 2;

  char* p = (char*)d_ws;
  auto carve = [&](size_t bytes) -> void* {
    void* q = (void*)p;
    p += (bytes + 255) & ~(size_t)255;
    return q;
  };
  float*  maskf = (float*)carve((size_t)DHW * 4);
  float*  m2    = (float*)carve((size_t)D2 * HW * 4);
  float*  m3    = (float*)carve((size_t)D3 * HW * 4);
  float*  bns   = (float*)carve(320 * 4);
  float*  bnt   = (float*)carve(320 * 4);
  bf16_t* xb    = (bf16_t*)carve((size_t)DHW * 128 * 2);  // dead after layer 1
  bf16_t* h1    = (bf16_t*)carve((size_t)DHW * 64 * 2);
  bf16_t* h2    = (bf16_t*)carve((size_t)D2 * HW * 64 * 2);
  bf16_t* wt1   = (bf16_t*)carve((size_t)27 * 64 * 128 * 2);
  bf16_t* wt2   = (bf16_t*)carve((size_t)3 * 64 * 64 * 2);
  bf16_t* wt3   = (bf16_t*)carve((size_t)27 * 64 * 64 * 2);
  bf16_t* wt4   = (bf16_t*)carve((size_t)27 * 64 * 64 * 2);
  bf16_t* wt5   = (bf16_t*)carve((size_t)3 * 64 * 64 * 2);
  // h3/h4 alias the xb region (free after layer 1; 2 * 11.3M elems < 49.5M elems)
  bf16_t* h3 = xb;
  bf16_t* h4 = xb + (size_t)D2 * HW * 64;

  int n;
  n = DHW;      k_mask_thresh<<<(n + 255) / 256, 256, 0, stream>>>(mask_u, maskf, n);
  n = D2 * HW;  k_mask_down  <<<(n + 255) / 256, 256, 0, stream>>>(maskf, m2, D2, HW);
  n = D3 * HW;  k_mask_down  <<<(n + 255) / 256, 256, 0, stream>>>(m2, m3, D3, HW);
  k_bnfold<<<5, 64, 0, stream>>>(bng, bnb, bnm, bnv, bns, bnt);

  k_x_to_dhwc<<<dim3(W / 16, D * H, C / 16), dim3(16, 16), 0, stream>>>(x, maskf, xb, C, W);

  n = 27 * 64 * 128; k_wrepack<<<(n + 255) / 256, 256, 0, stream>>>(w1, wt1, 64, 128, 27);
  n = 3 * 64 * 64;   k_wrepack<<<(n + 255) / 256, 256, 0, stream>>>(w2, wt2, 64, 64, 3);
  n = 27 * 64 * 64;  k_wrepack<<<(n + 255) / 256, 256, 0, stream>>>(w3, wt3, 64, 64, 27);
  n = 27 * 64 * 64;  k_wrepack<<<(n + 255) / 256, 256, 0, stream>>>(w4, wt4, 64, 64, 27);
  n = 3 * 64 * 64;   k_wrepack<<<(n + 255) / 256, 256, 0, stream>>>(w5, wt5, 64, 64, 3);

  // L1: subm1 128->64 (D=11)
  k_conv_wmma<128, true ><<<dim3(W / 16, H / 2, D),  32, 0, stream>>>(
      xb, wt1, bns + 0,   bnt + 0,   maskf, h1, nullptr, D,  D,  H, W);
  // L2: sconv1 (3,1,1)/(2,1,1): 11 -> 5
  k_conv_wmma<64,  false><<<dim3(W / 16, H / 2, D2), 32, 0, stream>>>(
      h1, wt2, bns + 64,  bnt + 64,  m2,    h2, nullptr, D,  D2, H, W);
  // L3: subm2
  k_conv_wmma<64,  true ><<<dim3(W / 16, H / 2, D2), 32, 0, stream>>>(
      h2, wt3, bns + 128, bnt + 128, m2,    h3, nullptr, D2, D2, H, W);
  // L4: subm3
  k_conv_wmma<64,  true ><<<dim3(W / 16, H / 2, D2), 32, 0, stream>>>(
      h3, wt4, bns + 192, bnt + 192, m2,    h4, nullptr, D2, D2, H, W);
  // L5: sconv2: 5 -> 2, final fp32 NCDHW output
  k_conv_wmma<64,  false><<<dim3(W / 16, H / 2, D3), 32, 0, stream>>>(
      h4, wt5, bns + 256, bnt + 256, m3,    nullptr, out, D2, D3, H, W);
}
